// CausalSelfAttention_83416854822972
// MI455X (gfx1250) — compile-verified, matrix operand order checked by reading, not run on hardware
//
#include <hip/hip_runtime.h>
#include <hip/hip_bf16.h>

#define T_SEQ 2048
#define BATCH 4
#define C_DIM 768
#define NH 12
#define HD 64
#define M_TOK (BATCH * T_SEQ)   // 8192

typedef float v8f __attribute__((ext_vector_type(8)));
typedef __bf16 bf16_t;
typedef bf16_t v16bf __attribute__((ext_vector_type(16)));
typedef int v4i_t __attribute__((ext_vector_type(4)));

union FragBF { v16bf v; unsigned int u[8]; };

#if defined(__has_builtin)
#if __has_builtin(__builtin_amdgcn_global_load_async_to_lds_b128)
#define HAVE_ASYNC_LDS 1
#endif
#endif

// 16B global -> LDS copy. Async (ASYNCcnt-tracked, bypasses VGPRs) when the
// toolchain exposes the gfx1250 builtin; synchronous fallback otherwise.
__device__ __forceinline__ void cp_b128(void* lds, const void* g) {
#ifdef HAVE_ASYNC_LDS
  __builtin_amdgcn_global_load_async_to_lds_b128(
      (__attribute__((address_space(1))) v4i_t*)(g),
      (__attribute__((address_space(3))) v4i_t*)(lds), 0, 0);
#else
  *(uint4*)lds = *(const uint4*)g;
#endif
}
__device__ __forceinline__ void cp_b128_wait() {
#ifdef HAVE_ASYNC_LDS
  asm volatile("s_wait_asynccnt 0" ::: "memory");
#endif
}

__device__ __forceinline__ unsigned short f32_to_bf16(float f) {
  union { float f; unsigned int u; } x; x.f = f;
  unsigned int r = x.u + 0x7FFFu + ((x.u >> 16) & 1u);
  return (unsigned short)(r >> 16);
}

__global__ void cvt_f32_bf16_kernel(const float* __restrict__ in,
                                    unsigned short* __restrict__ out, int n) {
  int i = blockIdx.x * blockDim.x + threadIdx.x;
  if (i < n) out[i] = f32_to_bf16(in[i]);
}

// Load a 16x32 bf16 A/B fragment chunk from LDS.
// rowPtr points at the start of this row's 32-element K-chunk (as uints,
// 16 uints per chunk). ISA layout: VGPR j<4 holds K=8*half+2j,+1 ;
// VGPR j>=4 holds K=16+8*half+2(j-4),+1  => two contiguous 16B runs.
__device__ __forceinline__ v16bf load_frag(const unsigned int* rowPtr, int half) {
  FragBF f;
  *(uint4*)&f.u[0] = *(const uint4*)(rowPtr + 4 * half);
  *(uint4*)&f.u[4] = *(const uint4*)(rowPtr + 8 + 4 * half);
  return f.v;
}

// C[M,N] = A[M,K](bf16,row-major) @ B[K,N](bf16,row-major) + bias
// Block: 256 threads = 8 waves in a 4(m) x 2(n) grid; wave tile 32x32.
// Block tile: 128(M) x 64(N), K-step 64 (two WMMA k-chunks per LDS phase).
template <bool OUT_BF16>
__global__ __launch_bounds__(256) void gemm_bias_kernel(
    const unsigned short* __restrict__ A, const unsigned short* __restrict__ B,
    const float* __restrict__ bias, void* __restrict__ Cout,
    int M, int N, int K) {
  __shared__ unsigned int sA[128 * 32];  // [128 rows][64 bf16]
  __shared__ unsigned int sB[64 * 32];   // [64 n][64 k]  (transposed)

  const int tid = threadIdx.x;
  const int lane = tid & 31;
  const int wave = tid >> 5;
  const int wm = wave >> 1;       // 0..3
  const int wn = wave & 1;        // 0..1
  const int half = lane >> 4;
  const int ln = lane & 15;

  const int m0 = blockIdx.x * 128;
  const int n0 = blockIdx.y * 64;

  v8f acc[2][2];
#pragma unroll
  for (int mi = 0; mi < 2; ++mi)
#pragma unroll
    for (int ni = 0; ni < 2; ++ni) acc[mi][ni] = {};

  for (int k0 = 0; k0 < K; k0 += 64) {
    __syncthreads();
    // A tile: 128x64 bf16 = 1024 uint4; 4 per thread (async -> LDS)
#pragma unroll
    for (int i = 0; i < 4; ++i) {
      int idx = tid + i * 256;
      int row = idx >> 3;   // 8 uint4 per 64-bf16 row
      int cc = idx & 7;
      cp_b128((uint4*)sA + row * 8 + cc,
              A + (size_t)(m0 + row) * K + k0 + cc * 8);
    }
    // B tile: 64(k)x64(n), stored transposed [n][k]
#pragma unroll
    for (int i = 0; i < 2; ++i) {
      int idx = tid + i * 256;
      int k = idx >> 3;           // 0..63
      int nc = (idx & 7) * 8;
      uint4 d4 = *(const uint4*)(B + (size_t)(k0 + k) * N + n0 + nc);
      const unsigned short* e = (const unsigned short*)&d4;
      unsigned short* sBs = (unsigned short*)sB;
#pragma unroll
      for (int j = 0; j < 8; ++j) sBs[(nc + j) * 64 + k] = e[j];
    }
    cp_b128_wait();
    __syncthreads();

#pragma unroll
    for (int kk = 0; kk < 2; ++kk) {
      v16bf afrag[2], bfrag[2];
#pragma unroll
      for (int mi = 0; mi < 2; ++mi)
        afrag[mi] = load_frag(sA + (wm * 32 + mi * 16 + ln) * 32 + kk * 16, half);
#pragma unroll
      for (int ni = 0; ni < 2; ++ni)
        bfrag[ni] = load_frag(sB + (wn * 32 + ni * 16 + ln) * 32 + kk * 16, half);
#pragma unroll
      for (int mi = 0; mi < 2; ++mi)
#pragma unroll
        for (int ni = 0; ni < 2; ++ni)
          acc[mi][ni] = __builtin_amdgcn_wmma_f32_16x16x32_bf16(
              false, afrag[mi], false, bfrag[ni], (short)0, acc[mi][ni], false,
              false);
    }
  }

#pragma unroll
  for (int mi = 0; mi < 2; ++mi)
#pragma unroll
    for (int ni = 0; ni < 2; ++ni) {
      int mbase = m0 + wm * 32 + mi * 16 + 8 * half;
      int ncol = n0 + wn * 32 + ni * 16 + ln;
      float bv = bias[ncol];
#pragma unroll
      for (int r = 0; r < 8; ++r) {
        float val = acc[mi][ni][r] + bv;
        if (OUT_BF16)
          ((unsigned short*)Cout)[(size_t)(mbase + r) * N + ncol] =
              f32_to_bf16(val);
        else
          ((float*)Cout)[(size_t)(mbase + r) * N + ncol] = val;
      }
    }
}

// Flash attention. qkv: [M_TOK][2304] bf16 (q|k|v). y: [M_TOK][768] bf16.
// Grid: (T/64, B*H). Block: 128 threads = 4 waves; each wave = 16 q rows.
__global__ __launch_bounds__(128) void flash_attn_kernel(
    const unsigned short* __restrict__ qkv, unsigned short* __restrict__ y) {
  __shared__ unsigned int sQ[64 * 32];   // [64 q][64 d]
  __shared__ unsigned int sK[64 * 32];   // [64 key][64 d]
  __shared__ unsigned int sVT[64 * 32];  // [64 d][64 key] (transposed)
  __shared__ unsigned int sP[4 * 16 * 32];  // per wave: [16 q][64 key]

  const int tid = threadIdx.x;
  const int lane = tid & 31;
  const int wave = tid >> 5;
  const int half = lane >> 4;
  const int ln = lane & 15;

  const int q0 = blockIdx.x * 64;
  const int bh = blockIdx.y;
  const int b = bh / NH, h = bh % NH;
  const size_t rs = 3 * C_DIM;  // 2304
  const unsigned short* qB = qkv + (size_t)b * T_SEQ * rs + h * HD;
  const unsigned short* kB = qB + C_DIM;
  const unsigned short* vB = qB + 2 * C_DIM;

  // Load Q tile (64x64 bf16), async -> LDS
#pragma unroll
  for (int i = 0; i < 4; ++i) {
    int idx = tid + i * 128;
    int row = idx >> 3, cc = idx & 7;
    cp_b128((uint4*)sQ + row * 8 + cc,
            qB + (size_t)(q0 + row) * rs + cc * 8);
  }

  v8f oacc[4];
#pragma unroll
  for (int dt = 0; dt < 4; ++dt) oacc[dt] = {};
  float mrow[8], lrow[8];
#pragma unroll
  for (int r = 0; r < 8; ++r) { mrow[r] = -1e30f; lrow[r] = 0.f; }
  const float scale = 0.125f;  // 1/sqrt(64)

  const int nChunks = q0 / 64 + 1;
  for (int c = 0; c < nChunks; ++c) {
    const int c0 = c * 64;
    __syncthreads();
    // K chunk (row-major, async) and V chunk (transposed into sVT)
#pragma unroll
    for (int i = 0; i < 4; ++i) {
      int idx = tid + i * 128;
      int row = idx >> 3, cc = idx & 7;
      cp_b128((uint4*)sK + row * 8 + cc,
              kB + (size_t)(c0 + row) * rs + cc * 8);
      uint4 d4 = *(const uint4*)(vB + (size_t)(c0 + row) * rs + cc * 8);
      const unsigned short* e = (const unsigned short*)&d4;
      unsigned short* sVs = (unsigned short*)sVT;
#pragma unroll
      for (int j = 0; j < 8; ++j) sVs[(cc * 8 + j) * 64 + row] = e[j];
    }
    cp_b128_wait();
    __syncthreads();

    // ---- S = (Q K^T) * scale, causal-masked ----
    v8f sacc[4];
#pragma unroll
    for (int nt = 0; nt < 4; ++nt) sacc[nt] = {};
    v16bf aq[2];
    aq[0] = load_frag(sQ + (wave * 16 + ln) * 32, half);
    aq[1] = load_frag(sQ + (wave * 16 + ln) * 32 + 16, half);
#pragma unroll
    for (int nt = 0; nt < 4; ++nt)
#pragma unroll
      for (int kk = 0; kk < 2; ++kk) {
        v16bf bk = load_frag(sK + (nt * 16 + ln) * 32 + kk * 16, half);
        sacc[nt] = __builtin_amdgcn_wmma_f32_16x16x32_bf16(
            false, aq[kk], false, bk, (short)0, sacc[nt], false, false);
      }

    // ---- online softmax ----
    float mnew[8];
#pragma unroll
    for (int r = 0; r < 8; ++r) {
      int qrow = q0 + wave * 16 + r + 8 * half;
      float mx = mrow[r];
#pragma unroll
      for (int nt = 0; nt < 4; ++nt) {
        int key = c0 + nt * 16 + ln;
        float s = sacc[nt][r] * scale;
        s = (key <= qrow) ? s : -1e30f;
        sacc[nt][r] = s;
        mx = fmaxf(mx, s);
      }
      mx = fmaxf(mx, __shfl_xor(mx, 1, 16));
      mx = fmaxf(mx, __shfl_xor(mx, 2, 16));
      mx = fmaxf(mx, __shfl_xor(mx, 4, 16));
      mx = fmaxf(mx, __shfl_xor(mx, 8, 16));
      mnew[r] = mx;
    }
    unsigned short* sPw = (unsigned short*)sP + wave * 16 * 64;
#pragma unroll
    for (int r = 0; r < 8; ++r) {
      float alpha = __expf(mrow[r] - mnew[r]);
      float psum = 0.f;
#pragma unroll
      for (int nt = 0; nt < 4; ++nt) {
        float p = __expf(sacc[nt][r] - mnew[r]);
        psum += p;
        sPw[(r + 8 * half) * 64 + nt * 16 + ln] = f32_to_bf16(p);
      }
      psum += __shfl_xor(psum, 1, 16);
      psum += __shfl_xor(psum, 2, 16);
      psum += __shfl_xor(psum, 4, 16);
      psum += __shfl_xor(psum, 8, 16);
      lrow[r] = lrow[r] * alpha + psum;
      mrow[r] = mnew[r];
#pragma unroll
      for (int dt = 0; dt < 4; ++dt) oacc[dt][r] *= alpha;
    }
    // P staged in wave-private LDS: ensure writes land before ds_load frags
    asm volatile("s_wait_dscnt 0" ::: "memory");

    // ---- O += P @ V ----
    const unsigned int* sPu = sP + wave * 16 * 32;
    v16bf ap[2];
    ap[0] = load_frag(sPu + ln * 32, half);
    ap[1] = load_frag(sPu + ln * 32 + 16, half);
#pragma unroll
    for (int dt = 0; dt < 4; ++dt)
#pragma unroll
      for (int kk = 0; kk < 2; ++kk) {
        v16bf bv = load_frag(sVT + (dt * 16 + ln) * 32 + kk * 16, half);
        oacc[dt] = __builtin_amdgcn_wmma_f32_16x16x32_bf16(
            false, ap[kk], false, bv, (short)0, oacc[dt], false, false);
      }
  }

  // normalize + store (bf16, layout [token][h*64+d])
  unsigned short* yB = y + (size_t)b * T_SEQ * C_DIM + h * HD;
#pragma unroll
  for (int r = 0; r < 8; ++r) {
    float inv = 1.0f / lrow[r];
    int row = q0 + wave * 16 + r + 8 * half;
#pragma unroll
    for (int dt = 0; dt < 4; ++dt)
      yB[(size_t)row * C_DIM + dt * 16 + ln] = f32_to_bf16(oacc[dt][r] * inv);
  }
}

extern "C" void kernel_launch(void* const* d_in, const int* in_sizes, int n_in,
                              void* d_out, int out_size, void* d_ws,
                              size_t ws_size, hipStream_t stream) {
  const float* x = (const float*)d_in[0];
  const float* w_attn = (const float*)d_in[1];
  const float* b_attn = (const float*)d_in[2];
  const float* w_proj = (const float*)d_in[3];
  const float* b_proj = (const float*)d_in[4];
  float* out = (float*)d_out;

  char* ws = (char*)d_ws;
  size_t off = 0;
  auto alloc = [&](size_t bytes) {
    void* p = ws + off;
    off = (off + bytes + 255) & ~(size_t)255;
    return p;
  };
  unsigned short* xb  = (unsigned short*)alloc((size_t)M_TOK * C_DIM * 2);
  unsigned short* wab = (unsigned short*)alloc((size_t)C_DIM * 3 * C_DIM * 2);
  unsigned short* wpb = (unsigned short*)alloc((size_t)C_DIM * C_DIM * 2);
  unsigned short* qkvb = (unsigned short*)alloc((size_t)M_TOK * 3 * C_DIM * 2);
  unsigned short* yb = xb;  // reuse: x consumed by QKV GEMM before attn runs

  int n1 = M_TOK * C_DIM;
  cvt_f32_bf16_kernel<<<(n1 + 255) / 256, 256, 0, stream>>>(x, xb, n1);
  int n2 = C_DIM * 3 * C_DIM;
  cvt_f32_bf16_kernel<<<(n2 + 255) / 256, 256, 0, stream>>>(w_attn, wab, n2);
  int n3 = C_DIM * C_DIM;
  cvt_f32_bf16_kernel<<<(n3 + 255) / 256, 256, 0, stream>>>(w_proj, wpb, n3);

  dim3 g1(M_TOK / 128, (3 * C_DIM) / 64);
  gemm_bias_kernel<true><<<g1, 256, 0, stream>>>(xb, wab, b_attn, qkvb, M_TOK,
                                                 3 * C_DIM, C_DIM);

  dim3 g2(T_SEQ / 64, BATCH * NH);
  flash_attn_kernel<<<g2, 128, 0, stream>>>(qkvb, yb);

  dim3 g3(M_TOK / 128, C_DIM / 64);
  gemm_bias_kernel<false><<<g3, 256, 0, stream>>>(yb, wpb, b_proj, out, M_TOK,
                                                  C_DIM, C_DIM);
}
